// UNBlocks_35708358099272
// MI455X (gfx1250) — matrix operand order repaired and checked by reading, not compiled, not run on hardware
//
#include <hip/hip_runtime.h>
#include <hip/hip_bf16.h>

typedef _Float16 v8h  __attribute__((ext_vector_type(8)));
typedef _Float16 v16h __attribute__((ext_vector_type(16)));
typedef float    v8f  __attribute__((ext_vector_type(8)));

#define NPTS   300000
#define KOFF   27
#define CIN1   64
#define COUTC  96
#define BN_EPS 1e-5f

// ---------------------------------------------------------------- helpers ---

__global__ void f32_to_f16_kernel(const float* __restrict__ src,
                                  _Float16* __restrict__ dst, long long n) {
  long long i = (long long)blockIdx.x * blockDim.x + threadIdx.x;
  long long stride = (long long)gridDim.x * blockDim.x;
  for (; i < n; i += stride) dst[i] = (_Float16)src[i];
}

__global__ void zero_kernel(float* __restrict__ p, int n) {
  int i = blockIdx.x * blockDim.x + threadIdx.x;
  if (i < n) p[i] = 0.0f;
}

// --------------------------------------------------- sparse conv via WMMA ---
// One wave32 computes a 32-row x 96-col f32 output tile (two 16-row M-tiles
// sharing every B fragment -> halves B bandwidth per FLOP):
//   out[i, :] = sum_k  xh[nbrs[i,k], :] @ W[k]     (f16 A/B, f32 accumulate)
// A fragment (16x32 f16): lane l -> row (l&15), K = 8*(l>>4) + 16*(e>>3) + (e&7)
// B fragment (32x16 f16): lane l -> N = (l&15), element e -> K = 8*(l>>4) + 16*(e>>3) + (e&7)  (row-major W!)
// C/D fragment: VGPR r, lane l -> M = r + 8*(l>>4), N = (l&15)
template<int CIN>
__global__ void __launch_bounds__(256)
sparse_conv_wmma(const _Float16* __restrict__ xh,
                 const int* __restrict__ nbrs,
                 const _Float16* __restrict__ wh,
                 float* __restrict__ out, int nTiles) {
  const int wave = blockIdx.x * (blockDim.x >> 5) + (threadIdx.x >> 5);
  if (wave >= nTiles) return;                 // wave-uniform: EXEC stays all-1s
  const int lane   = threadIdx.x & 31;
  const int rowIn  = lane & 15;
  const int half   = lane >> 4;
  const int rowBase = wave * 32;

  v8f acc0[6] = {};                           // rows rowBase +  0..15
  v8f acc1[6] = {};                           // rows rowBase + 16..31

  const int* __restrict__ nrow0 = nbrs + (size_t)(rowBase + rowIn) * KOFF;
  const int* __restrict__ nrow1 = nbrs + (size_t)(rowBase + 16 + rowIn) * KOFF;

  for (int k = 0; k < KOFF; ++k) {
    const int idx0 = nrow0[k];
    const int idx1 = nrow1[k];
    const _Float16* __restrict__ arow0 = xh + (size_t)idx0 * CIN + 8 * half;
    const _Float16* __restrict__ arow1 = xh + (size_t)idx1 * CIN + 8 * half;
#pragma unroll
    for (int c = 0; c < CIN / 32; ++c) {
      // gather the two 16x32 A tiles for this cin-chunk
      v8h lo0 = *(const v8h*)(arow0 + 32 * c);
      v8h hi0 = *(const v8h*)(arow0 + 32 * c + 16);
      v8h lo1 = *(const v8h*)(arow1 + 32 * c);
      v8h hi1 = *(const v8h*)(arow1 + 32 * c + 16);
      v16h a0 = __builtin_shufflevector(lo0, hi0, 0, 1, 2, 3, 4, 5, 6, 7,
                                                  8, 9, 10, 11, 12, 13, 14, 15);
      v16h a1 = __builtin_shufflevector(lo1, hi1, 0, 1, 2, 3, 4, 5, 6, 7,
                                                  8, 9, 10, 11, 12, 13, 14, 15);
      // B: lane l supplies W[k][32c + 8*half + {0..7,16..23}][16t + (l&15)]
      const _Float16* __restrict__ brow =
          wh + ((size_t)(k * CIN + 32 * c + 8 * half)) * COUTC + rowIn;
#pragma unroll
      for (int t = 0; t < 6; ++t) {
        v16h b;
#pragma unroll
        for (int s = 0; s < 16; ++s)
          b[s] = brow[(size_t)(s + (s & 8)) * COUTC + 16 * t];
        acc0[t] = __builtin_amdgcn_wmma_f32_16x16x32_f16(
            false, a0, false, b, (short)0, acc0[t], false, false);
        acc1[t] = __builtin_amdgcn_wmma_f32_16x16x32_f16(
            false, a1, false, b, (short)0, acc1[t], false, false);
      }
    }
  }

  float* __restrict__ orow0 = out + (size_t)rowBase * COUTC;
  float* __restrict__ orow1 = out + (size_t)(rowBase + 16) * COUTC;
#pragma unroll
  for (int t = 0; t < 6; ++t)
#pragma unroll
    for (int r = 0; r < 8; ++r) {
      orow0[(size_t)(r + 8 * half) * COUTC + 16 * t + rowIn] = acc0[t][r];
      orow1[(size_t)(r + 8 * half) * COUTC + 16 * t + rowIn] = acc1[t][r];
    }
}

// ------------------------------------------------------------- batch norm ---

__global__ void __launch_bounds__(256)
bn_stats_kernel(const float* __restrict__ h, float* __restrict__ sum,
                float* __restrict__ sq, long long total) {
  __shared__ float ls[COUTC], lq[COUTC];
  if (threadIdx.x < COUTC) { ls[threadIdx.x] = 0.f; lq[threadIdx.x] = 0.f; }
  __syncthreads();
  long long i = (long long)blockIdx.x * blockDim.x + threadIdx.x;
  long long stride = (long long)gridDim.x * blockDim.x;
  for (; i < total; i += stride) {
    float v = h[i];
    int c = (int)(i % COUTC);
    atomicAdd(&ls[c], v);
    atomicAdd(&lq[c], v * v);
  }
  __syncthreads();
  if (threadIdx.x < COUTC) {
    atomicAdd(&sum[threadIdx.x], ls[threadIdx.x]);
    atomicAdd(&sq[threadIdx.x], lq[threadIdx.x]);
  }
}

// BN + ReLU, emit f16 (feeds the next conv's A-matrix path)
__global__ void __launch_bounds__(256)
bn_relu_to_f16(const float* __restrict__ h, const float* __restrict__ sum,
               const float* __restrict__ sq, const float* __restrict__ g,
               const float* __restrict__ beta, _Float16* __restrict__ o,
               long long total, float invN) {
  long long i = (long long)blockIdx.x * blockDim.x + threadIdx.x;
  long long stride = (long long)gridDim.x * blockDim.x;
  for (; i < total; i += stride) {
    int c = (int)(i % COUTC);
    float mu  = sum[c] * invN;
    float var = sq[c] * invN - mu * mu;
    float s   = rsqrtf(var + BN_EPS) * g[c];
    float v   = (h[i] - mu) * s + beta[c];
    o[i] = (_Float16)fmaxf(v, 0.0f);
  }
}

// BN + ReLU in place on f32 (final output)
__global__ void __launch_bounds__(256)
bn_relu_inplace_f32(float* h, const float* __restrict__ sum,
                    const float* __restrict__ sq, const float* __restrict__ g,
                    const float* __restrict__ beta, long long total, float invN) {
  long long i = (long long)blockIdx.x * blockDim.x + threadIdx.x;
  long long stride = (long long)gridDim.x * blockDim.x;
  for (; i < total; i += stride) {
    int c = (int)(i % COUTC);
    float mu  = sum[c] * invN;
    float var = sq[c] * invN - mu * mu;
    float s   = rsqrtf(var + BN_EPS) * g[c];
    float v   = (h[i] - mu) * s + beta[c];
    h[i] = fmaxf(v, 0.0f);
  }
}

// ----------------------------------------------------------------- launch ---

extern "C" void kernel_launch(void* const* d_in, const int* in_sizes, int n_in,
                              void* d_out, int out_size, void* d_ws, size_t ws_size,
                              hipStream_t stream) {
  const float* x    = (const float*)d_in[0];   // [N, 64]
  const int*   nbrs = (const int*)  d_in[1];   // [N, 27]
  const float* W1   = (const float*)d_in[2];   // [27, 64, 96]
  const float* g1   = (const float*)d_in[3];
  const float* b1   = (const float*)d_in[4];
  const float* W2   = (const float*)d_in[5];   // [27, 96, 96]
  const float* g2   = (const float*)d_in[6];
  const float* b2   = (const float*)d_in[7];
  float* out = (float*)d_out;                  // [N, 96], also used as h2 scratch

  const long long nX  = (long long)NPTS * CIN1;          // 19,200,000
  const long long nW1 = (long long)KOFF * CIN1 * COUTC;  //    165,888
  const long long nW2 = (long long)KOFF * COUTC * COUTC; //    248,832
  const long long nH  = (long long)NPTS * COUTC;         // 28,800,000

  // workspace layout (all offsets 256B-aligned)
  char* ws = (char*)d_ws;
  _Float16* xh   = (_Float16*)(ws);                         // 38,400,000 B
  _Float16* w1h  = (_Float16*)(ws + 38400000);              //    331,776 B
  _Float16* w2h  = (_Float16*)(ws + 38731776);              //    497,664 B
  float*    h1   = (float*)   (ws + 39229440);              // 115,200,000 B
  _Float16* h1h  = (_Float16*)(ws + 154429440);             // 57,600,000 B
  float*    stats = (float*)  (ws + 212029440);             // 4*96 floats
  float* sum1 = stats;        float* sq1 = stats + COUTC;
  float* sum2 = stats + 192;  float* sq2 = stats + 288;

  // 1) precision conversion (f32 -> f16) for x, W1, W2
  f32_to_f16_kernel<<<4096, 256, 0, stream>>>(x,  xh,  nX);
  f32_to_f16_kernel<<<648,  256, 0, stream>>>(W1, w1h, nW1);
  f32_to_f16_kernel<<<972,  256, 0, stream>>>(W2, w2h, nW2);
  zero_kernel<<<2, 256, 0, stream>>>(stats, 4 * COUTC);

  const int nTiles = NPTS / 32;                 // 9,375 (exact)
  const int convBlocks = (nTiles + 7) / 8;      // 8 waves/block

  // 2) conv1 (Cin=64) -> h1 (f32)
  sparse_conv_wmma<CIN1><<<convBlocks, 256, 0, stream>>>(xh, nbrs, w1h, h1, nTiles);

  // 3) BN1 stats + fused BN/ReLU -> f16
  bn_stats_kernel<<<2048, 256, 0, stream>>>(h1, sum1, sq1, nH);
  bn_relu_to_f16<<<4096, 256, 0, stream>>>(h1, sum1, sq1, g1, b1, h1h, nH,
                                           1.0f / (float)NPTS);

  // 4) conv2 (Cin=96) -> d_out used as raw f32 scratch
  sparse_conv_wmma<COUTC><<<convBlocks, 256, 0, stream>>>(h1h, nbrs, w2h, out, nTiles);

  // 5) BN2 stats + fused BN/ReLU in place on d_out
  bn_stats_kernel<<<2048, 256, 0, stream>>>(out, sum2, sq2, nH);
  bn_relu_inplace_f32<<<4096, 256, 0, stream>>>(out, sum2, sq2, g2, b2, nH,
                                                1.0f / (float)NPTS);
  (void)in_sizes; (void)n_in; (void)out_size; (void)ws_size;
}
